// GroupQueryAttention_28355374088581
// MI455X (gfx1250) — compile-verified, matrix operand order checked by reading, not run on hardware
//
#include <hip/hip_runtime.h>
#include <hip/hip_bf16.h>
#include <stdint.h>

#define DM   2048   // d_model
#define DKV  512    // kv width
#define NQH  32
#define NKVH 8
#define GRP  4
#define DH   64
#define BATCH 2
#define SEQ  2048
#define MROWS (BATCH*SEQ)   // 4096

typedef __attribute__((ext_vector_type(16))) __bf16 v16bf;
typedef __attribute__((ext_vector_type(8)))  __bf16 v8bf;
typedef __attribute__((ext_vector_type(8)))  float  v8f;

__device__ __forceinline__ __bf16 to_bf16(float x) {
    union { float f; uint32_t u; } c; c.f = x;
    uint32_t r = (c.u + 0x7FFFu + ((c.u >> 16) & 1u)) >> 16;
    union { unsigned short s; __bf16 b; } o; o.s = (unsigned short)r;
    return o.b;
}
__device__ __forceinline__ float bf2f(__bf16 b) {
    union { unsigned short s; __bf16 b; } i; i.b = b;
    union { uint32_t u; float f; } o; o.u = ((uint32_t)i.s) << 16;
    return o.f;
}

// Load one 16x32 bf16 WMMA fragment (A or B operand) from a row-major tile.
// Per ISA 7.12.2: lanes 0-15 hold row (lane&15), k = {0..7, 16..23};
// lanes 16-31 hold row (lane&15), k = {8..15, 24..31}.
__device__ __forceinline__ v16bf load_frag(const __bf16* base, int stride, int lane) {
    const __bf16* p = base + (size_t)(lane & 15) * stride + (lane >> 4) * 8;
    v8bf lo = *(const v8bf*)p;
    v8bf hi = *(const v8bf*)(p + 16);
    v16bf f;
#pragma unroll
    for (int i = 0; i < 8; ++i) { f[i] = lo[i]; f[i + 8] = hi[i]; }
    return f;
}

__device__ __forceinline__ v8f wmma_bf16(v16bf a, v16bf b, v8f c) {
    return __builtin_amdgcn_wmma_f32_16x16x32_bf16(false, a, false, b, (short)0, c,
                                                   false, false);
}

// ---------------------------------------------------------------- conversions
__global__ void cvt_f32_bf16(const float* __restrict__ in, __bf16* __restrict__ out,
                             int n) {
    int i = blockIdx.x * blockDim.x + threadIdx.x;
    int stride = gridDim.x * blockDim.x;
    for (; i < n; i += stride) out[i] = to_bf16(in[i]);
}

// -------------------------------------------------------------------- NT GEMM
// C[M,N] = A[M,K](row) * W[N,K](row)^T + bias[N]
// block tile 128x128, 8 waves in 2(M)x4(N), each wave 64x32 -> 4x2 WMMA tiles.
// K stepped by 64 through LDS (two WMMA k-steps per barrier pair).
template <bool OUT_BF16>
__global__ __launch_bounds__(256) void gemm_nt(const __bf16* __restrict__ A,
                                               const __bf16* __restrict__ W,
                                               const float* __restrict__ bias,
                                               void* __restrict__ out,
                                               int M, int N, int K) {
    __shared__ __bf16 Al[128 * 72];
    __shared__ __bf16 Bl[128 * 72];

    const int tid  = threadIdx.x;
    const int lane = tid & 31;
    const int wave = tid >> 5;
    const int wm   = wave >> 2;   // 0..1
    const int wn   = wave & 3;    // 0..3
    const int m0   = blockIdx.x * 128;
    const int n0   = blockIdx.y * 128;

    v8f acc[4][2];
#pragma unroll
    for (int i = 0; i < 4; ++i)
#pragma unroll
        for (int j = 0; j < 2; ++j)
            acc[i][j] = v8f{0.f, 0.f, 0.f, 0.f, 0.f, 0.f, 0.f, 0.f};

    const int lr = tid >> 1;          // 0..127 staging row
    const int lc = (tid & 1) * 32;    // 0 or 32 (each thread: 32 contiguous bf16)

    for (int k0 = 0; k0 < K; k0 += 64) {
        const __bf16* ag = A + (size_t)(m0 + lr) * K + k0 + lc;
        const __bf16* bg = W + (size_t)(n0 + lr) * K + k0 + lc;
#pragma unroll
        for (int c = 0; c < 32; c += 8) {
            *(v8bf*)&Al[lr * 72 + lc + c] = *(const v8bf*)(ag + c);
            *(v8bf*)&Bl[lr * 72 + lc + c] = *(const v8bf*)(bg + c);
        }
        if (k0 + 64 < K) {                       // warm L2/WGP$ for next step
            __builtin_prefetch(ag + 64, 0, 1);
            __builtin_prefetch(bg + 64, 0, 1);
        }
        __syncthreads();

#pragma unroll
        for (int ks = 0; ks < 64; ks += 32) {
            v16bf af[4], bf[2];
#pragma unroll
            for (int i = 0; i < 4; ++i)
                af[i] = load_frag(&Al[(wm * 64 + i * 16) * 72 + ks], 72, lane);
#pragma unroll
            for (int j = 0; j < 2; ++j)
                bf[j] = load_frag(&Bl[(wn * 32 + j * 16) * 72 + ks], 72, lane);
#pragma unroll
            for (int i = 0; i < 4; ++i)
#pragma unroll
                for (int j = 0; j < 2; ++j)
                    acc[i][j] = wmma_bf16(af[i], bf[j], acc[i][j]);
        }
        __syncthreads();
    }

    // epilogue: C layout -> lane<16: M=v, lane>=16: M=v+8; col = lane&15
    const int mb  = m0 + wm * 64 + (lane >> 4) * 8;
    const int cb  = n0 + wn * 32 + (lane & 15);
#pragma unroll
    for (int j = 0; j < 2; ++j) {
        const int c  = cb + j * 16;
        const float bb = bias[c];
#pragma unroll
        for (int i = 0; i < 4; ++i) {
            const int rbase = mb + i * 16;
#pragma unroll
            for (int v = 0; v < 8; ++v) {
                const float val = acc[i][j][v] + bb;
                const size_t idx = (size_t)(rbase + v) * N + c;
                if (OUT_BF16) ((__bf16*)out)[idx] = to_bf16(val);
                else          ((float*)out)[idx]  = val;
            }
        }
    }
}

// ------------------------------------------------------------- flash GQA attn
// grid: (SEQ/128, NQH, BATCH), 256 threads. Each wave: 16 query rows.
// 64-key chunks: 16 WMMAs per chunk against one fixed-cost softmax reduction.
__global__ __launch_bounds__(256) void flash_gqa(const __bf16* __restrict__ Q,
                                                 const __bf16* __restrict__ Kg,
                                                 const __bf16* __restrict__ Vg,
                                                 __bf16* __restrict__ O) {
    __shared__ __bf16 Kl[64 * 72];      // [key][d]   stride 72
    __shared__ __bf16 Vl[64 * 72];      // [d][key]   (transposed) stride 72
    __shared__ __bf16 Pl[8][16 * 72];   // per-wave P round-trip, stride 72

    const int tid  = threadIdx.x;
    const int lane = tid & 31;
    const int wave = tid >> 5;
    const int qh   = blockIdx.y;        // 0..31
    const int kvh  = qh >> 2;           // GQA: q head h uses kv head h/4
    const int b    = blockIdx.z;
    const int row0 = blockIdx.x * 128 + wave * 16;

    // preload Q fragments; fold in 1/sqrt(64)=0.125 (exact power of two in bf16)
    const __bf16* qbase = Q + ((size_t)b * SEQ + row0) * DM + qh * DH;
    v16bf qf0 = load_frag(qbase, DM, lane);
    v16bf qf1 = load_frag(qbase + 32, DM, lane);
#pragma unroll
    for (int i = 0; i < 16; ++i) {
        qf0[i] = to_bf16(bf2f(qf0[i]) * 0.125f);
        qf1[i] = to_bf16(bf2f(qf1[i]) * 0.125f);
    }

    const v8f vzero = v8f{0.f, 0.f, 0.f, 0.f, 0.f, 0.f, 0.f, 0.f};
    v8f o[4] = {vzero, vzero, vzero, vzero};
    float rm[8], rl[8];
#pragma unroll
    for (int v = 0; v < 8; ++v) { rm[v] = -1e30f; rl[v] = 0.f; }

    const int skey = tid >> 2;          // staging: 0..63 keys
    const int sc   = (tid & 3) * 16;    // 0,16,32,48 within d

    for (int kc = 0; kc < SEQ; kc += 64) {
        // stage K chunk [64 x 64] row-major
        const __bf16* kg = Kg + ((size_t)b * SEQ + kc + skey) * DKV + kvh * DH + sc;
        const __bf16* vg = Vg + ((size_t)b * SEQ + kc + skey) * DKV + kvh * DH + sc;
        *(v8bf*)&Kl[skey * 72 + sc]     = *(const v8bf*)kg;
        *(v8bf*)&Kl[skey * 72 + sc + 8] = *(const v8bf*)(kg + 8);
        // stage V chunk transposed -> Vl[d][key]
        v8bf v0 = *(const v8bf*)vg;
        v8bf v1 = *(const v8bf*)(vg + 8);
#pragma unroll
        for (int e = 0; e < 8; ++e) {
            Vl[(sc + e) * 72 + skey]     = v0[e];
            Vl[(sc + e + 8) * 72 + skey] = v1[e];
        }
        __syncthreads();

        // S[16 x 64] = (Q/sqrt(d)) K^T : 4 key-tiles x 2 d-steps = 8 WMMAs
        v8f s[4] = {vzero, vzero, vzero, vzero};
#pragma unroll
        for (int t = 0; t < 4; ++t) {
            s[t] = wmma_bf16(qf0, load_frag(&Kl[(t * 16) * 72], 72, lane), s[t]);
            s[t] = wmma_bf16(qf1, load_frag(&Kl[(t * 16) * 72 + 32], 72, lane), s[t]);
        }

        // online softmax (rows live in 16-lane halves -> xor masks 1..8)
        float p[4][8], corr[8];
#pragma unroll
        for (int v = 0; v < 8; ++v) {
            float mx = fmaxf(fmaxf(s[0][v], s[1][v]), fmaxf(s[2][v], s[3][v]));
#pragma unroll
            for (int msk = 1; msk < 16; msk <<= 1)
                mx = fmaxf(mx, __shfl_xor(mx, msk));
            const float nm = fmaxf(rm[v], mx);
            corr[v] = __expf(rm[v] - nm);
            float rs = 0.f;
#pragma unroll
            for (int t = 0; t < 4; ++t) {
                p[t][v] = __expf(s[t][v] - nm);
                rs += p[t][v];
            }
#pragma unroll
            for (int msk = 1; msk < 16; msk <<= 1)
                rs += __shfl_xor(rs, msk);
            rl[v] = rl[v] * corr[v] + rs;
            rm[v] = nm;
        }
#pragma unroll
        for (int t = 0; t < 4; ++t)
#pragma unroll
            for (int v = 0; v < 8; ++v) o[t][v] *= corr[v];

        // P: C-layout -> LDS -> A-fragment layout (per-wave buffer)
        __bf16* pw = &Pl[wave][0];
        const int prow = (lane >> 4) * 8;
        const int pcol = lane & 15;
#pragma unroll
        for (int t = 0; t < 4; ++t)
#pragma unroll
            for (int v = 0; v < 8; ++v)
                pw[(prow + v) * 72 + t * 16 + pcol] = to_bf16(p[t][v]);
        asm volatile("s_wait_dscnt 0" ::: "memory");
        const v16bf pf0 = load_frag(pw, 72, lane);        // keys 0..31
        const v16bf pf1 = load_frag(pw + 32, 72, lane);   // keys 32..63

        // O += P * V : 4 d-tiles x 2 key-steps = 8 WMMAs (keys contiguous in Vl)
#pragma unroll
        for (int t = 0; t < 4; ++t) {
            o[t] = wmma_bf16(pf0, load_frag(&Vl[(t * 16) * 72], 72, lane), o[t]);
            o[t] = wmma_bf16(pf1, load_frag(&Vl[(t * 16) * 72 + 32], 72, lane), o[t]);
        }
        __syncthreads();
    }

    // epilogue: normalize and store bf16 into [B,N,DM] at this head's columns
    const int prow = (lane >> 4) * 8;
    const int pcol = lane & 15;
#pragma unroll
    for (int v = 0; v < 8; ++v) {
        const float inv = 1.0f / rl[v];
        const size_t base =
            ((size_t)b * SEQ + row0 + prow + v) * DM + qh * DH + pcol;
#pragma unroll
        for (int t = 0; t < 4; ++t)
            O[base + t * 16] = to_bf16(o[t][v] * inv);
    }
}

// --------------------------------------------------------------------- launch
extern "C" void kernel_launch(void* const* d_in, const int* in_sizes, int n_in,
                              void* d_out, int out_size, void* d_ws, size_t ws_size,
                              hipStream_t stream) {
    const float* x  = (const float*)d_in[0];
    const float* Wq = (const float*)d_in[1];
    const float* bq = (const float*)d_in[2];
    const float* Wk = (const float*)d_in[3];
    const float* bk = (const float*)d_in[4];
    const float* Wv = (const float*)d_in[5];
    const float* bv = (const float*)d_in[6];
    const float* Wo = (const float*)d_in[7];
    const float* bo = (const float*)d_in[8];
    float* out = (float*)d_out;

    char* ws = (char*)d_ws;
    size_t off = 0;
    auto alloc = [&](size_t elems) {
        __bf16* p = (__bf16*)(ws + off);
        off += ((elems * 2 + 255) & ~(size_t)255);
        return p;
    };
    __bf16* xb  = alloc((size_t)MROWS * DM);
    __bf16* wqb = alloc((size_t)DM * DM);
    __bf16* wkb = alloc((size_t)DKV * DM);
    __bf16* wvb = alloc((size_t)DKV * DM);
    __bf16* wob = alloc((size_t)DM * DM);
    __bf16* Qb  = alloc((size_t)MROWS * DM);
    __bf16* Kb  = alloc((size_t)MROWS * DKV);
    __bf16* Vb  = alloc((size_t)MROWS * DKV);
    __bf16* Ob  = alloc((size_t)MROWS * DM);
    (void)ws_size; (void)in_sizes; (void)n_in; (void)out_size;

    // 1) f32 -> bf16 conversions
    cvt_f32_bf16<<<4096, 256, 0, stream>>>(x,  xb,  MROWS * DM);
    cvt_f32_bf16<<<2048, 256, 0, stream>>>(Wq, wqb, DM * DM);
    cvt_f32_bf16<<<1024, 256, 0, stream>>>(Wk, wkb, DKV * DM);
    cvt_f32_bf16<<<1024, 256, 0, stream>>>(Wv, wvb, DKV * DM);
    cvt_f32_bf16<<<2048, 256, 0, stream>>>(Wo, wob, DM * DM);

    // 2) projections: Q = x Wq^T + bq ; K = x Wk^T + bk ; V = x Wv^T + bv
    gemm_nt<true><<<dim3(MROWS / 128, DM / 128), 256, 0, stream>>>(
        xb, wqb, bq, Qb, MROWS, DM, DM);
    gemm_nt<true><<<dim3(MROWS / 128, DKV / 128), 256, 0, stream>>>(
        xb, wkb, bk, Kb, MROWS, DKV, DM);
    gemm_nt<true><<<dim3(MROWS / 128, DKV / 128), 256, 0, stream>>>(
        xb, wvb, bv, Vb, MROWS, DKV, DM);

    // 3) flash attention with grouped KV heads
    flash_gqa<<<dim3(SEQ / 128, NQH, BATCH), 256, 0, stream>>>(Qb, Kb, Vb, Ob);

    // 4) output projection (f32 result straight to d_out)
    gemm_nt<false><<<dim3(MROWS / 128, DM / 128), 256, 0, stream>>>(
        Ob, wob, bo, out, MROWS, DM, DM);
}
